// CrossAttention_11948599017473
// MI455X (gfx1250) — matrix operand order repaired and checked by reading, not compiled, not run on hardware
//
#include <hip/hip_runtime.h>
#include <hip/hip_bf16.h>
#include <math.h>

// Problem constants (from reference): B=32, N=4096, C=1024, H=16, DH=64
#define BB 32
#define NN 4096
#define CC 1024
#define HH 16
#define DH 64
#define SCALE 0.125f   // DH^-0.5

typedef __attribute__((ext_vector_type(16))) _Float16 v16h;
typedef __attribute__((ext_vector_type(8)))  _Float16 v8h;
typedef __attribute__((ext_vector_type(8)))  float    v8f;

// ---------------------------------------------------------------------------
// Kernel 1: q[b,j] = x[b,0,:] . wq[j,:]        (wq read exactly once)
// grid 8, block 128; each block owns 128 rows of wq, loops over b.
// ---------------------------------------------------------------------------
__global__ void k1_query(const float* __restrict__ x, const float* __restrict__ wq,
                         float* __restrict__ q) {
    __shared__ float xs[CC];
    const int t = threadIdx.x;
    const int j = blockIdx.x * 128 + t;
    const float* wrow = wq + (size_t)j * CC;
    for (int b = 0; b < BB; ++b) {
        for (int i = t; i < CC; i += 128) xs[i] = x[(size_t)b * NN * CC + i];
        __syncthreads();
        float acc = 0.f;
        #pragma unroll 4
        for (int c = 0; c < CC; c += 4) {
            float4 w4 = *(const float4*)(wrow + c);
            acc += w4.x * xs[c] + w4.y * xs[c + 1] + w4.z * xs[c + 2] + w4.w * xs[c + 3];
        }
        q[b * CC + j] = acc;
        __syncthreads();
    }
}

// ---------------------------------------------------------------------------
// Kernel 2: qk[b,h,c] = SCALE * sum_d q[b,h*64+d] * wk[h*64+d, c]
// grid (16 h, 8 ctile), block 128; wk slice staged in LDS, read once total.
// ---------------------------------------------------------------------------
__global__ void k2_foldk(const float* __restrict__ q, const float* __restrict__ wk,
                         float* __restrict__ qk) {
    __shared__ float wks[DH * 128];   // 32 KB
    __shared__ float qs[DH];
    const int h = blockIdx.x, c0 = blockIdx.y * 128, t = threadIdx.x;
    for (int e = t; e < DH * 128; e += 128) {
        int d = e >> 7, cc = e & 127;
        wks[e] = wk[(size_t)(h * DH + d) * CC + c0 + cc];
    }
    __syncthreads();
    for (int b = 0; b < BB; ++b) {
        if (t < DH) qs[t] = q[b * CC + h * DH + t];
        __syncthreads();
        float acc = 0.f;
        #pragma unroll 8
        for (int d = 0; d < DH; ++d) acc += qs[d] * wks[d * 128 + t];
        qk[((size_t)b * HH + h) * CC + c0 + t] = acc * SCALE;
        __syncthreads();
    }
}

// ---------------------------------------------------------------------------
// Kernel 3: logits[b,n,h] = x[b,n,:] . qk[b,h,:]  - 1000*(1-mask[b,n])
// Main x-streaming pass #1.  WMMA f32_16x16x32_f16.
//   A (16 n-rows x 32 k): ISA A layout = lane&15 is the M row; lanes 0-15
//     hold K kb..kb+7 (V0-3) and kb+16..kb+23 (V4-7) with kb = 0/8 by lane half.
//   B (32 k x 16 heads): ISA B layout = lane&15 is the N col; each lane holds
//     K kb..kb+7 and kb+16..kb+23 (same order as A)  -> two 16B LDS
//     vector loads per k-step (row stride padded to 1032 halves = 516 dwords,
//     516 % 64 == 4, so lanes 0-15 hit distinct banks).
// grid (32 ntiles, 32 b), block 256 = 8 waves, 16 rows/wave.
// ---------------------------------------------------------------------------
#define QKS 1032
__global__ void k3_logits(const float* __restrict__ x, const float* __restrict__ qk,
                          const float* __restrict__ mask, float* __restrict__ logits) {
    __shared__ _Float16 qs[HH * QKS];   // ~33 KB
    const int b  = blockIdx.y;
    const int n0 = blockIdx.x * 128;
    const int t  = threadIdx.x;
    for (int e = t; e < HH * CC; e += 256) {
        int h = e >> 10, c = e & (CC - 1);
        qs[h * QKS + c] = (_Float16)qk[((size_t)b * HH + h) * CC + c];
    }
    __syncthreads();

    const int wave = t >> 5, lane = t & 31;
    const int ln15 = lane & 15;
    const int kbA  = (lane & 16) ? 8 : 0;    // A operand K sub-block
    const int kbB  = (lane & 16) ? 16 : 0;   // unused; B is loaded with kbA, as A is
    const int row  = n0 + wave * 16 + ln15;
    const float*    xrow = x  + ((size_t)b * NN + row) * CC;
    const _Float16* qrow = qs + ln15 * QKS;

    v8f acc = {};
    for (int k0 = 0; k0 < CC; k0 += 32) {
        float4 x0 = *(const float4*)(xrow + k0 + kbA);
        float4 x1 = *(const float4*)(xrow + k0 + kbA + 4);
        float4 x2 = *(const float4*)(xrow + k0 + kbA + 16);
        float4 x3 = *(const float4*)(xrow + k0 + kbA + 20);
        v16h a;
        a[0]=(_Float16)x0.x; a[1]=(_Float16)x0.y; a[2]=(_Float16)x0.z; a[3]=(_Float16)x0.w;
        a[4]=(_Float16)x1.x; a[5]=(_Float16)x1.y; a[6]=(_Float16)x1.z; a[7]=(_Float16)x1.w;
        a[8]=(_Float16)x2.x; a[9]=(_Float16)x2.y; a[10]=(_Float16)x2.z; a[11]=(_Float16)x2.w;
        a[12]=(_Float16)x3.x; a[13]=(_Float16)x3.y; a[14]=(_Float16)x3.z; a[15]=(_Float16)x3.w;
        // B: K kbA..kbA+7 and kbA+16..kbA+23 (matches A), 16B-aligned in LDS -> two ds_load_b128
        const v8h* bp = (const v8h*)(qrow + k0 + kbA);
        v8h b0 = bp[0], b1 = bp[2];
        v16h bv = __builtin_shufflevector(b0, b1, 0,1,2,3,4,5,6,7,8,9,10,11,12,13,14,15);
        acc = __builtin_amdgcn_wmma_f32_16x16x32_f16(false, a, false, bv,
                                                     (short)0, acc, false, false);
    }
    // C layout: VGPR r -> M = (lane&16 ? 8:0) + r, N = lane&15.  Store [b][n][h].
    const int mOff  = kbA;               // 0 or 8, same as C-layout M offset
    const int rbase = n0 + wave * 16 + mOff;
    float4 m0 = *(const float4*)(mask + (size_t)b * NN + rbase);
    float4 m1 = *(const float4*)(mask + (size_t)b * NN + rbase + 4);
    float mk[8] = {m0.x, m0.y, m0.z, m0.w, m1.x, m1.y, m1.z, m1.w};
    #pragma unroll
    for (int r = 0; r < 8; ++r) {
        int n = rbase + r;
        logits[((size_t)b * NN + n) * HH + ln15] = acc[r] - 1000.0f * (1.0f - mk[r]);
    }
}

// ---------------------------------------------------------------------------
// Kernel 4: in-place softmax over n for each (b,h).  grid (16 h, 32 b), 256 thr.
// ---------------------------------------------------------------------------
__global__ void k4_softmax(float* __restrict__ logits) {
    __shared__ float red[256];
    const int b = blockIdx.y, h = blockIdx.x, t = threadIdx.x;
    float* base = logits + (size_t)b * NN * HH + h;
    float lv[16];
    float m = -1e30f;
    #pragma unroll
    for (int i = 0; i < 16; ++i) {
        lv[i] = base[(size_t)(i * 256 + t) * HH];
        m = fmaxf(m, lv[i]);
    }
    red[t] = m; __syncthreads();
    for (int s = 128; s > 0; s >>= 1) { if (t < s) red[t] = fmaxf(red[t], red[t + s]); __syncthreads(); }
    m = red[0]; __syncthreads();
    float sum = 0.f;
    #pragma unroll
    for (int i = 0; i < 16; ++i) { lv[i] = __expf(lv[i] - m); sum += lv[i]; }
    red[t] = sum; __syncthreads();
    for (int s = 128; s > 0; s >>= 1) { if (t < s) red[t] += red[t + s]; __syncthreads(); }
    const float inv = 1.0f / red[0];
    #pragma unroll
    for (int i = 0; i < 16; ++i) base[(size_t)(i * 256 + t) * HH] = lv[i] * inv;
}

// ---------------------------------------------------------------------------
// Kernel 5: weighted[b,c,h] = sum_n attn[b,n,h] * x[b,n,c]
// Main x-streaming pass #2.  A (16 channels x 32 n) from x columns
// (lane-coalesced across channels), B (32 n x 16 heads) from attn (L2),
// filled in the same K order as A.  grid (64 ctiles, 32 b),
// block 256 = 8 waves splitting K=4096; LDS tree reduction.
// ---------------------------------------------------------------------------
__global__ void k5_weighted(const float* __restrict__ x, const float* __restrict__ attn,
                            float* __restrict__ weighted) {
    __shared__ float red[8][256];   // 8 KB
    const int b = blockIdx.y, c0 = blockIdx.x * 16, t = threadIdx.x;
    const int wave = t >> 5, lane = t & 31;
    const int ln15 = lane & 15;       // channel row of A, head col of B
    const int kbA = (lane & 16) ? 8 : 0;
    const int kbB = (lane & 16) ? 16 : 0;
    const float* xb = x    + (size_t)b * NN * CC;
    const float* ab = attn + (size_t)b * NN * HH;

    v8f acc = {};
    const int kend = wave * 512 + 512;
    for (int k0 = wave * 512; k0 < kend; k0 += 32) {
        v16h a, bv;
        #pragma unroll
        for (int i = 0; i < 8; ++i) {
            int n1 = k0 + kbA + i, n2 = k0 + kbA + 16 + i;
            a[i]     = (_Float16)xb[(size_t)n1 * CC + c0 + ln15];
            a[8 + i] = (_Float16)xb[(size_t)n2 * CC + c0 + ln15];
        }
        #pragma unroll
        for (int i = 0; i < 16; ++i) {
            bv[i] = (_Float16)ab[(size_t)(k0 + kbA + i + (i >= 8 ? 8 : 0)) * HH + ln15];
        }
        acc = __builtin_amdgcn_wmma_f32_16x16x32_f16(false, a, false, bv,
                                                     (short)0, acc, false, false);
    }
    #pragma unroll
    for (int r = 0; r < 8; ++r) red[wave][(kbA + r) * 16 + ln15] = acc[r];
    __syncthreads();
    // 256 threads: one (M,N) cell each, sum 8 wave partials
    const int M = t >> 4, Nh = t & 15;
    float s = 0.f;
    #pragma unroll
    for (int w = 0; w < 8; ++w) s += red[w][M * 16 + Nh];
    weighted[(size_t)b * CC * HH + (size_t)(c0 + M) * HH + Nh] = s;
}

// ---------------------------------------------------------------------------
// Kernel 6a: oh[b,j] = weighted[b, :, j>>6] . wv[j, :]   (wv read once)
// grid 8, block 128 (block owns 128 j-rows = exactly 2 heads), loop over b.
// ---------------------------------------------------------------------------
__global__ void k6a_headproj(const float* __restrict__ weighted, const float* __restrict__ wv,
                             float* __restrict__ oh) {
    __shared__ float ws[2][CC];   // 8 KB
    const int t = threadIdx.x, j0 = blockIdx.x * 128, h0 = j0 >> 6;
    const int j = j0 + t, hh = t >> 6;
    const float* wrow = wv + (size_t)j * CC;
    for (int b = 0; b < BB; ++b) {
        const float* wb = weighted + (size_t)b * CC * HH;
        for (int e = t; e < 2 * CC; e += 128) {
            int hl = e >> 10, c = e & (CC - 1);
            ws[hl][c] = wb[(size_t)c * HH + h0 + hl];
        }
        __syncthreads();
        float acc = 0.f;
        #pragma unroll 4
        for (int c = 0; c < CC; c += 4) {
            float4 w4 = *(const float4*)(wrow + c);
            acc += w4.x * ws[hh][c] + w4.y * ws[hh][c+1] + w4.z * ws[hh][c+2] + w4.w * ws[hh][c+3];
        }
        oh[b * CC + j] = acc;
        __syncthreads();
    }
}

// ---------------------------------------------------------------------------
// Kernel 6b: out[b,j] = proj_b[j] + oh[b,:] . proj_w[j,:]  (proj_w read once)
// ---------------------------------------------------------------------------
__global__ void k6b_outproj(const float* __restrict__ oh, const float* __restrict__ pw,
                            const float* __restrict__ pb, float* __restrict__ out) {
    __shared__ float os[CC];
    const int t = threadIdx.x, j = blockIdx.x * 128 + t;
    const float* prow = pw + (size_t)j * CC;
    const float bias = pb[j];
    for (int b = 0; b < BB; ++b) {
        for (int e = t; e < CC; e += 128) os[e] = oh[b * CC + e];
        __syncthreads();
        float acc = bias;
        #pragma unroll 4
        for (int c = 0; c < CC; c += 4) {
            float4 w4 = *(const float4*)(prow + c);
            acc += w4.x * os[c] + w4.y * os[c+1] + w4.z * os[c+2] + w4.w * os[c+3];
        }
        out[b * CC + j] = acc;
        __syncthreads();
    }
}

// ---------------------------------------------------------------------------
extern "C" void kernel_launch(void* const* d_in, const int* in_sizes, int n_in,
                              void* d_out, int out_size, void* d_ws, size_t ws_size,
                              hipStream_t stream) {
    (void)in_sizes; (void)n_in; (void)out_size; (void)ws_size;
    const float* x      = (const float*)d_in[0];
    const float* mask   = (const float*)d_in[1];
    const float* wq     = (const float*)d_in[2];
    const float* wk     = (const float*)d_in[3];
    const float* wv     = (const float*)d_in[4];
    const float* proj_w = (const float*)d_in[5];
    const float* proj_b = (const float*)d_in[6];
    float* out = (float*)d_out;

    // Workspace layout (floats): q | qk | logits(in-place attn) | weighted | oh
    float* ws       = (float*)d_ws;
    float* q        = ws;                               // 32*1024
    float* qk       = q  + BB * CC;                     // 32*16*1024
    float* logits   = qk + (size_t)BB * HH * CC;        // 32*4096*16
    float* weighted = logits + (size_t)BB * NN * HH;    // 32*1024*16
    float* oh       = weighted + (size_t)BB * CC * HH;  // 32*1024

    k1_query  <<<dim3(8),        dim3(128), 0, stream>>>(x, wq, q);
    k2_foldk  <<<dim3(16, 8),    dim3(128), 0, stream>>>(q, wk, qk);
    k3_logits <<<dim3(32, 32),   dim3(256), 0, stream>>>(x, qk, mask, logits);
    k4_softmax<<<dim3(16, 32),   dim3(256), 0, stream>>>(logits);
    k5_weighted<<<dim3(64, 32),  dim3(256), 0, stream>>>(x, logits, weighted);
    k6a_headproj<<<dim3(8),      dim3(128), 0, stream>>>(weighted, wv, oh);
    k6b_outproj <<<dim3(8),      dim3(128), 0, stream>>>(oh, proj_w, proj_b, out);
}
